// EDM_Block_75024488726860
// MI455X (gfx1250) — matrix operand order repaired and checked by reading, not compiled, not run on hardware
//
#include <hip/hip_runtime.h>
#include <hip/hip_bf16.h>

#define HDIM   128
#define NNODES 40000
#define NEDGES 400000

typedef __attribute__((ext_vector_type(16))) _Float16 v16h;
typedef __attribute__((ext_vector_type(8)))  float    v8f;

union HV16 { v16h v; float4 f4[2]; };

// A operand: lane holds row m=(lane&15); halves 0..7 = K[abase..abase+7],
// halves 8..15 = K[abase+16..abase+23], abase = (lane<16)?0:8.
__device__ __forceinline__ v16h load_a16(const _Float16* p) {
    HV16 u;
    u.f4[0] = *reinterpret_cast<const float4*>(p);
    u.f4[1] = *reinterpret_cast<const float4*>(p + 16);
    return u.v;
}
// B operand (pre-packed): 16 contiguous halves per lane.
__device__ __forceinline__ v16h load_b16(const _Float16* p) {
    HV16 u;
    u.f4[0] = *reinterpret_cast<const float4*>(p);
    u.f4[1] = *reinterpret_cast<const float4*>(p + 8);
    return u.v;
}

// ---------------------------------------------------------------------------
// Pack W (KxN fp32 row-major) into WMMA B-operand tiles:
// layout [nt][kt][lane][16 halves]; lane: n = nt*16+(lane&15),
// K = kt*32 + ((lane<16)?0:8) + j + (j>=8 ? 8 : 0), j=0..15.
// ---------------------------------------------------------------------------
__global__ void pack_weights(const float* __restrict__ W, _Float16* __restrict__ Wp,
                             int K, int N) {
    int idx = blockIdx.x * blockDim.x + threadIdx.x;
    int KT = K >> 5, NT = N >> 4;
    int total = NT * KT * 32;
    if (idx >= total) return;
    int lane = idx & 31;
    int t = idx >> 5;
    int kt = t % KT;
    int nt = t / KT;
    int n  = (nt << 4) + (lane & 15);
    int kb = (kt << 5) + ((lane < 16) ? 0 : 8);
    _Float16* o = Wp + (size_t)idx * 16;
#pragma unroll
    for (int j = 0; j < 16; ++j)
        o[j] = (_Float16)W[(size_t)(kb + j + ((j >> 3) << 3)) * N + n];
}

// ---------------------------------------------------------------------------
// N-register-blocked WMMA GEMM: each wave computes 16 x (NB*16) of C.
// Per K-step: A loaded once, all NB B tiles loaded up front (one clause,
// partial waits), then NB independent WMMAs (no accumulator RAW chain).
// MODE 0: out_f16 = (relu?) result
// MODE 1: bond:   v *= bond[etype[row]*H+n]; out_f32 & out_f16
// MODE 2: conv:   v = (relu?) v; v += add[row]; out_f32
// ---------------------------------------------------------------------------
template <int NB, int MODE>
__global__ void gemm_wmma_nb(const _Float16* __restrict__ A,
                             const _Float16* __restrict__ Wp,
                             const float* __restrict__ bias,
                             int M, int K, int NT, int relu_flag,
                             float* __restrict__ out_f32,
                             _Float16* __restrict__ out_f16,
                             const float* __restrict__ add_f32,
                             const float* __restrict__ bond_table,
                             const int* __restrict__ etype) {
    int lane = threadIdx.x & 31;
    int wave = blockIdx.x * (blockDim.x >> 5) + (threadIdx.x >> 5);
    int ngroups = NT / NB;
    int total = (M >> 4) * ngroups;
    if (wave >= total) return;              // wave-uniform
    int mt = wave / ngroups;
    int nt0 = (wave - mt * ngroups) * NB;
    int m = (mt << 4) + (lane & 15);
    int abase = (lane < 16) ? 0 : 8;
    int KT = K >> 5;
    int N = NT << 4;
    const _Float16* arow = A  + (size_t)m * K + abase;
    const _Float16* wp   = Wp + ((size_t)nt0 * KT * 512) + (size_t)lane * 16;
    v8f acc[NB];
#pragma unroll
    for (int nb = 0; nb < NB; ++nb) acc[nb] = (v8f){};
    for (int kt = 0; kt < KT; ++kt) {
        v16h a = load_a16(arow + (kt << 5));
        const _Float16* wk = wp + (size_t)kt * 512;
        v16h b[NB];
#pragma unroll
        for (int nb = 0; nb < NB; ++nb)
            b[nb] = load_b16(wk + (size_t)nb * KT * 512);
#pragma unroll
        for (int nb = 0; nb < NB; ++nb)
            acc[nb] = __builtin_amdgcn_wmma_f32_16x16x32_f16(false, a, false, b[nb],
                                                             (short)0, acc[nb], false, false);
    }
    int mb = (lane < 16) ? 0 : 8;
#pragma unroll
    for (int nb = 0; nb < NB; ++nb) {
        int n = ((nt0 + nb) << 4) + (lane & 15);
        float bv = bias[n];
#pragma unroll
        for (int r = 0; r < 8; ++r) {
            int row = (mt << 4) + mb + r;
            size_t o = (size_t)row * N + n;
            float v = acc[nb][r] + bv;
            if (MODE == 0) {
                if (relu_flag) v = fmaxf(v, 0.f);
                out_f16[o] = (_Float16)v;
            } else if (MODE == 1) {
                v *= bond_table[(size_t)etype[row] * HDIM + n];
                out_f32[o] = v;
                out_f16[o] = (_Float16)v;
            } else {
                if (relu_flag) v = fmaxf(v, 0.f);
                v += add_f32[o];
                out_f32[o] = v;
            }
        }
    }
}

// ---------------------------------------------------------------------------
// Gather-GEMM for h_pair = [na[src], na[dst], ea]  (E x 384) @ (384 x 128).
// One wave per 16-edge tile computes the full 16x128 output row block, so
// the random-access row gathers happen exactly once per edge.
// ---------------------------------------------------------------------------
__global__ void gemm_gather_wmma(const _Float16* __restrict__ na,
                                 const _Float16* __restrict__ ea,
                                 const int* __restrict__ src,
                                 const int* __restrict__ dst,
                                 const _Float16* __restrict__ Wp,
                                 const float* __restrict__ bias,
                                 int E, _Float16* __restrict__ out) {
    const int NB = 8, KT = 12, N = 128;
    int lane = threadIdx.x & 31;
    int wave = blockIdx.x * (blockDim.x >> 5) + (threadIdx.x >> 5);
    int total = E >> 4;
    if (wave >= total) return;
    int mt = wave;
    int e = (mt << 4) + (lane & 15);
    int abase = (lane < 16) ? 0 : 8;
    const _Float16* s0 = na + (size_t)src[e] * HDIM + abase;
    const _Float16* s1 = na + (size_t)dst[e] * HDIM + abase;
    const _Float16* s2 = ea + (size_t)e      * HDIM + abase;
    const _Float16* wp = Wp + (size_t)lane * 16;
    v8f acc[NB];
#pragma unroll
    for (int nb = 0; nb < NB; ++nb) acc[nb] = (v8f){};
#pragma unroll
    for (int kt = 0; kt < KT; ++kt) {
        const _Float16* ap = (kt < 4) ? s0 : ((kt < 8) ? s1 : s2);
        v16h a = load_a16(ap + ((kt & 3) << 5));
        const _Float16* wk = wp + (size_t)kt * 512;
        v16h b[NB];
#pragma unroll
        for (int nb = 0; nb < NB; ++nb)
            b[nb] = load_b16(wk + (size_t)nb * KT * 512);
#pragma unroll
        for (int nb = 0; nb < NB; ++nb)
            acc[nb] = __builtin_amdgcn_wmma_f32_16x16x32_f16(false, a, false, b[nb],
                                                             (short)0, acc[nb], false, false);
    }
    int mb = (lane < 16) ? 0 : 8;
#pragma unroll
    for (int nb = 0; nb < NB; ++nb) {
        int n = (nb << 4) + (lane & 15);
        float bv = bias[n];
#pragma unroll
        for (int r = 0; r < 8; ++r) {
            int row = (mt << 4) + mb + r;
            float v = fmaxf(acc[nb][r] + bv, 0.f);
            out[(size_t)row * N + n] = (_Float16)v;
        }
    }
}

// ---------------------------------------------------------------------------
// Elementwise / scatter kernels (fp32 paths)
// ---------------------------------------------------------------------------
__global__ void edge_l1_kernel(const float* __restrict__ len,
                               const float* __restrict__ W1,
                               const float* __restrict__ b1,
                               _Float16* __restrict__ h1, int total) {
    int i = blockIdx.x * blockDim.x + threadIdx.x;
    if (i >= total) return;
    int e = i >> 7, h = i & 127;
    float v = len[e] * W1[h] + b1[h];
    h1[i] = (_Float16)fmaxf(v, 0.f);
}

__global__ void x_init_kernel(const float* __restrict__ table,
                              const int* __restrict__ deg,
                              float* __restrict__ x, int total) {
    int i = blockIdx.x * blockDim.x + threadIdx.x;
    if (i >= total) return;
    int node = i >> 7, h = i & 127;
    x[i] = table[(size_t)deg[node] * HDIM + h];
}

__global__ void msg_aggr_kernel(const float* __restrict__ x,
                                const float* __restrict__ ea,
                                const int* __restrict__ src,
                                const int* __restrict__ dst,
                                float* __restrict__ aggr, int E) {
    int idx = blockIdx.x * blockDim.x + threadIdx.x;
    if (idx >= E * 32) return;
    int e = idx >> 5, c = idx & 31;
    int s = src[e], d = dst[e];
    float4 xv = reinterpret_cast<const float4*>(x  + (size_t)s * HDIM)[c];
    float4 ev = reinterpret_cast<const float4*>(ea + (size_t)e * HDIM)[c];
    float m0 = fmaxf(xv.x + ev.x, 0.f);
    float m1 = fmaxf(xv.y + ev.y, 0.f);
    float m2 = fmaxf(xv.z + ev.z, 0.f);
    float m3 = fmaxf(xv.w + ev.w, 0.f);
    float* ap = aggr + (size_t)d * HDIM + c * 4;
    atomicAdd(ap + 0, m0);
    atomicAdd(ap + 1, m1);
    atomicAdd(ap + 2, m2);
    atomicAdd(ap + 3, m3);
}

__global__ void add_cvt_kernel(const float* __restrict__ x,
                               const float* __restrict__ aggr,
                               _Float16* __restrict__ out, int total) {
    int i = blockIdx.x * blockDim.x + threadIdx.x;
    if (i >= total) return;
    out[i] = (_Float16)(x[i] + aggr[i]);
}

__global__ void cvt_f16_kernel(const float* __restrict__ x,
                               _Float16* __restrict__ out, int total) {
    int i = blockIdx.x * blockDim.x + threadIdx.x;
    if (i >= total) return;
    out[i] = (_Float16)x[i];
}

__global__ void out_init_kernel(const float* __restrict__ pos,
                                float* __restrict__ out, int total) {
    int i = blockIdx.x * blockDim.x + threadIdx.x;
    if (i >= total) return;
    out[i] = pos[i];
}

// final layer of grad MLP (64->1 dot) + eq_transform scatter
__global__ void edge_final_kernel(const _Float16* __restrict__ y2,
                                  const float* __restrict__ gW3,
                                  const float* __restrict__ gb3,
                                  const float* __restrict__ pos,
                                  const float* __restrict__ len,
                                  const int* __restrict__ src,
                                  const int* __restrict__ dst,
                                  float* __restrict__ out, int E) {
    int e = blockIdx.x * blockDim.x + threadIdx.x;
    if (e >= E) return;
    const _Float16* yp = y2 + (size_t)e * 64;
    float acc = gb3[0];
#pragma unroll 16
    for (int j = 0; j < 64; ++j) acc += (float)yp[j] * gW3[j];
    int s = src[e], d = dst[e];
    float invl = 1.f / len[e];
    float sx = (pos[2 * s + 0] - pos[2 * d + 0]) * invl * acc;
    float sy = (pos[2 * s + 1] - pos[2 * d + 1]) * invl * acc;
    atomicAdd(&out[2 * s + 0], sx);
    atomicAdd(&out[2 * s + 1], sy);
    atomicAdd(&out[2 * d + 0], -sx);
    atomicAdd(&out[2 * d + 1], -sy);
}

// ---------------------------------------------------------------------------
// Host launcher
// ---------------------------------------------------------------------------
extern "C" void kernel_launch(void* const* d_in, const int* in_sizes, int n_in,
                              void* d_out, int out_size, void* d_ws, size_t ws_size,
                              hipStream_t stream) {
    // ----- input index resolution (insertion order vs jax sorted-key flatten)
    int iNodeDeg, iPos, iEdgeIdx, iEdgeType, iEdgeLen;
    int iDeg, iBond, iEW1, iEW2, iEB1, iEB2;
    int iCW1[4], iCB1[4], iCW2[4], iCB2[4];
    int iGW1, iGW2, iGW3, iGB1, iGB2, iGB3;

    bool sortedTop = (n_in >= 2 && in_sizes[0] == NNODES && in_sizes[1] == 2 * NEDGES);
    if (!sortedTop) {
        // top-level insertion order
        iNodeDeg = 2; iPos = 3; iEdgeIdx = 4; iEdgeType = 5; iEdgeLen = 6;
        bool sortedParams = (n_in > 10 && in_sizes[10] != 12800);
        if (!sortedParams) {
            iDeg = 9; iBond = 10; iEW1 = 11; iEW2 = 12; iEB1 = 13; iEB2 = 14;
            int q = 15;
            for (int i = 0; i < 4; ++i) { iCW1[i] = q++; iCB1[i] = q++; iCW2[i] = q++; iCB2[i] = q++; }
            iGW1 = 31; iGW2 = 32; iGW3 = 33; iGB1 = 34; iGB2 = 35; iGB3 = 36;
        } else {
            iBond = 9;
            int q = 10;
            for (int i = 0; i < 4; ++i) { iCW1[i] = q++; iCW2[i] = q++; iCB1[i] = q++; iCB2[i] = q++; }
            iDeg = 26; iEW1 = 27; iEW2 = 28; iEB1 = 29; iEB2 = 30;
            iGW1 = 31; iGW2 = 32; iGW3 = 33; iGB1 = 34; iGB2 = 35; iGB3 = 36;
        }
    } else {
        // fully sorted: batch, edge_index, edge_length, edge_type, node_degree,
        // node_emb, node_type, [params sorted], pos, time_step
        iEdgeIdx = 1; iEdgeLen = 2; iEdgeType = 3; iNodeDeg = 4; iPos = n_in - 2;
        iBond = 7;
        int q = 8;
        for (int i = 0; i < 4; ++i) { iCW1[i] = q++; iCW2[i] = q++; iCB1[i] = q++; iCB2[i] = q++; }
        iDeg = 24; iEW1 = 25; iEW2 = 26; iEB1 = 27; iEB2 = 28;
        iGW1 = 29; iGW2 = 30; iGW3 = 31; iGB1 = 32; iGB2 = 33; iGB3 = 34;
    }

    const int*   node_deg  = (const int*)d_in[iNodeDeg];
    const float* pos       = (const float*)d_in[iPos];
    const int*   src       = (const int*)d_in[iEdgeIdx];
    const int*   dst       = src + NEDGES;
    const int*   etype     = (const int*)d_in[iEdgeType];
    const float* elen      = (const float*)d_in[iEdgeLen];
    const float* deg_table = (const float*)d_in[iDeg];
    const float* bond_tab  = (const float*)d_in[iBond];
    const float* eW1 = (const float*)d_in[iEW1];
    const float* eW2 = (const float*)d_in[iEW2];
    const float* eb1 = (const float*)d_in[iEB1];
    const float* eb2 = (const float*)d_in[iEB2];
    const float* gW1 = (const float*)d_in[iGW1];
    const float* gW2 = (const float*)d_in[iGW2];
    const float* gW3 = (const float*)d_in[iGW3];
    const float* gb1 = (const float*)d_in[iGB1];
    const float* gb2 = (const float*)d_in[iGB2];
    const float* gb3 = (const float*)d_in[iGB3];
    float* out = (float*)d_out;

    // ----- workspace carve-up (256B aligned)
    char* base = (char*)d_ws;
    size_t off = 0;
    auto alloc = [&](size_t bytes) { char* p = base + off; off += (bytes + 255) & ~(size_t)255; return p; };
    _Float16* h1_f16  = (_Float16*)alloc((size_t)NEDGES * HDIM * 2); // also reused as y1
    float*    ea_f32  = (float*)   alloc((size_t)NEDGES * HDIM * 4);
    _Float16* ea_f16  = (_Float16*)alloc((size_t)NEDGES * HDIM * 2);
    float*    x_f32   = (float*)   alloc((size_t)NNODES * HDIM * 4);
    _Float16* xin_f16 = (_Float16*)alloc((size_t)NNODES * HDIM * 2);
    float*    aggr    = (float*)   alloc((size_t)NNODES * HDIM * 4);
    _Float16* hid_f16 = (_Float16*)alloc((size_t)NNODES * 2 * HDIM * 2);
    _Float16* y2_f16  = (_Float16*)alloc((size_t)NEDGES * 64 * 2);
    _Float16* eW2p    = (_Float16*)alloc((size_t)4 * 8 * 512 * 2);
    _Float16* cW1p[4], *cW2p[4];
    for (int i = 0; i < 4; ++i) {
        cW1p[i] = (_Float16*)alloc((size_t)4 * 16 * 512 * 2);
        cW2p[i] = (_Float16*)alloc((size_t)8 * 8 * 512 * 2);
    }
    _Float16* gW1p = (_Float16*)alloc((size_t)12 * 8 * 512 * 2);
    _Float16* gW2p = (_Float16*)alloc((size_t)4 * 4 * 512 * 2);
    _Float16* y1_f16 = h1_f16; // reuse after edge stage

    // ----- weight packing
    pack_weights<<<4,  256, 0, stream>>>(eW2, eW2p, 128, 128);
    for (int i = 0; i < 4; ++i) {
        pack_weights<<<8, 256, 0, stream>>>((const float*)d_in[iCW1[i]], cW1p[i], 128, 256);
        pack_weights<<<8, 256, 0, stream>>>((const float*)d_in[iCW2[i]], cW2p[i], 256, 128);
    }
    pack_weights<<<12, 256, 0, stream>>>(gW1, gW1p, 384, 128);
    pack_weights<<<2,  256, 0, stream>>>(gW2, gW2p, 128, 64);

    // ----- edge encoder
    edge_l1_kernel<<<(NEDGES * HDIM + 255) / 256, 256, 0, stream>>>(elen, eW1, eb1, h1_f16, NEDGES * HDIM);
    // d_emb = h1 @ eW2 + eb2; edge_attr = d_emb * bond[etype]   (25000 waves)
    gemm_wmma_nb<8, 1><<<6250, 128, 0, stream>>>(h1_f16, eW2p, eb2, NEDGES, 128, 8, 0,
                                                 ea_f32, ea_f16, nullptr, bond_tab, etype);

    // ----- node init
    x_init_kernel<<<(NNODES * HDIM + 255) / 256, 256, 0, stream>>>(deg_table, node_deg, x_f32, NNODES * HDIM);

    // ----- GINE conv layers
    for (int layer = 0; layer < 4; ++layer) {
        hipMemsetAsync(aggr, 0, (size_t)NNODES * HDIM * 4, stream);
        msg_aggr_kernel<<<(NEDGES * 32 + 255) / 256, 256, 0, stream>>>(x_f32, ea_f32, src, dst, aggr, NEDGES);
        add_cvt_kernel<<<(NNODES * HDIM + 255) / 256, 256, 0, stream>>>(x_f32, aggr, xin_f16, NNODES * HDIM);
        // hidden = relu((x+aggr) @ W1 + b1)  (N x 256): 2500 mtiles * 2 ngroups = 5000 waves
        gemm_wmma_nb<8, 0><<<1250, 128, 0, stream>>>(xin_f16, cW1p[layer], (const float*)d_in[iCB1[layer]],
                                                     NNODES, 128, 16, 1,
                                                     nullptr, hid_f16, nullptr, nullptr, nullptr);
        // x = (relu?)(hidden @ W2 + b2) + x : 2500 waves
        gemm_wmma_nb<8, 2><<<625, 128, 0, stream>>>(hid_f16, cW2p[layer], (const float*)d_in[iCB2[layer]],
                                                    NNODES, 256, 8, (layer < 3) ? 1 : 0,
                                                    x_f32, nullptr, x_f32, nullptr, nullptr);
    }
    // node_attr -> f16
    cvt_f16_kernel<<<(NNODES * HDIM + 255) / 256, 256, 0, stream>>>(x_f32, xin_f16, NNODES * HDIM);

    // ----- grad MLP
    // y1 = relu([na_src, na_dst, ea] @ gW1 + gb1) : 25000 waves
    gemm_gather_wmma<<<6250, 128, 0, stream>>>(xin_f16, ea_f16, src, dst, gW1p, gb1, NEDGES, y1_f16);
    // y2 = relu(y1 @ gW2 + gb2) : N=64 -> NT=4, NB=4 : 25000 waves
    gemm_wmma_nb<4, 0><<<6250, 128, 0, stream>>>(y1_f16, gW2p, gb2, NEDGES, 128, 4, 1,
                                                 nullptr, y2_f16, nullptr, nullptr, nullptr);

    // ----- eq_transform + output
    out_init_kernel<<<(2 * NNODES + 255) / 256, 256, 0, stream>>>(pos, out, 2 * NNODES);
    edge_final_kernel<<<(NEDGES + 255) / 256, 256, 0, stream>>>(y2_f16, gW3, gb3, pos, elen,
                                                                src, dst, out, NEDGES);
}
